// MoEFFN_75144747811320
// MI455X (gfx1250) — compile-verified, matrix operand order checked by reading, not run on hardware
//
#include <hip/hip_runtime.h>
#include <hip/hip_bf16.h>

typedef __attribute__((ext_vector_type(16))) __bf16 v16bf;
typedef __attribute__((ext_vector_type(8)))  __bf16 v8bf;
typedef __attribute__((ext_vector_type(2)))  __bf16 v2bf;
typedef __attribute__((ext_vector_type(8)))  float  v8f;

#define NT 8192      // tokens B*S
#define DIM 1024
#define HID 4096
#define NE 8
#define TM 64        // token tile
#define TN 128       // output-column tile
#define KS 32        // K step (bf16 wmma depth)
#define LDP 40       // LDS row stride elements (80B: 16B-aligned frags, bank-spread)

#define LD4(p) (*(const float4*)(p))

__device__ __forceinline__ unsigned pack_bf16(float a, float b) {
  v2bf t; t[0] = (__bf16)a; t[1] = (__bf16)b;
  return __builtin_bit_cast(unsigned, t);
}

__device__ __forceinline__ v16bf cat16(v8bf lo, v8bf hi) {
  return __builtin_shufflevector(lo, hi, 0,1,2,3,4,5,6,7,8,9,10,11,12,13,14,15);
}

// ---------------- Router ------------------------------------------------------
__global__ __launch_bounds__(256)
void moe_router(const float* __restrict__ x, const float* __restrict__ rw,
                const float* __restrict__ temp,
                int* __restrict__ topI, float* __restrict__ topW,
                float* __restrict__ blockImp, int* __restrict__ blockCnt,
                float* __restrict__ blockEnt) {
  const int tid = threadIdx.x;
  const int t = blockIdx.x * 256 + tid;
  const float tc = fminf(fmaxf(temp[0], 0.1f), 5.0f);

  float lg[NE];
#pragma unroll
  for (int e = 0; e < NE; ++e) lg[e] = 0.f;
  const float* xr = x + (size_t)t * DIM;
  for (int d = 0; d < DIM; ++d) {
    float xv = xr[d];
    const float4* r4 = (const float4*)(rw + d * NE);
    float4 a = r4[0], b = r4[1];
    lg[0] += xv * a.x; lg[1] += xv * a.y; lg[2] += xv * a.z; lg[3] += xv * a.w;
    lg[4] += xv * b.x; lg[5] += xv * b.y; lg[6] += xv * b.z; lg[7] += xv * b.w;
  }
#pragma unroll
  for (int e = 0; e < NE; ++e) lg[e] = lg[e] / tc;

  float mx = lg[0];
#pragma unroll
  for (int e = 1; e < NE; ++e) mx = fmaxf(mx, lg[e]);
  float p[NE]; float sum = 0.f;
#pragma unroll
  for (int e = 0; e < NE; ++e) { p[e] = __expf(lg[e] - mx); sum += p[e]; }
  float rinv = 1.f / sum;
#pragma unroll
  for (int e = 0; e < NE; ++e) p[e] *= rinv;

  int i0 = 0;
#pragma unroll
  for (int e = 1; e < NE; ++e) if (p[e] > p[i0]) i0 = e;
  int i1 = (i0 == 0) ? 1 : 0;
#pragma unroll
  for (int e = 0; e < NE; ++e) if (e != i0 && p[e] > p[i1]) i1 = e;

  topI[2 * t] = i0;  topI[2 * t + 1] = i1;
  topW[2 * t] = p[i0]; topW[2 * t + 1] = p[i1];

  float ent = 0.f;
#pragma unroll
  for (int e = 0; e < NE; ++e) ent -= p[e] * __logf(fmaxf(p[e], 1e-8f));

  __shared__ float red[256];
  __shared__ int ic[NE];
  for (int e = 0; e < NE; ++e) {
    red[tid] = p[e]; __syncthreads();
    for (int s = 128; s > 0; s >>= 1) { if (tid < s) red[tid] += red[tid + s]; __syncthreads(); }
    if (tid == 0) blockImp[blockIdx.x * NE + e] = red[0];
    __syncthreads();
  }
  if (tid < NE) ic[tid] = 0;
  __syncthreads();
  atomicAdd(&ic[i0], 1); atomicAdd(&ic[i1], 1);
  __syncthreads();
  if (tid < NE) blockCnt[blockIdx.x * NE + tid] = ic[tid];
  red[tid] = ent; __syncthreads();
  for (int s = 128; s > 0; s >>= 1) { if (tid < s) red[tid] += red[tid + s]; __syncthreads(); }
  if (tid == 0) blockEnt[blockIdx.x] = red[0];
}

// ---------------- Dispatch ----------------------------------------------------
__global__ __launch_bounds__(256)
void moe_dispatch(const int* __restrict__ topI, int* __restrict__ cnt,
                  int* __restrict__ lists) {
  const int t = blockIdx.x * 256 + threadIdx.x;
#pragma unroll
  for (int k = 0; k < 2; ++k) {
    int e = topI[2 * t + k];
    int pos = atomicAdd(&cnt[e], 1);
    lists[e * NT + pos] = 2 * t + k;
  }
}

// ---------------- Phase 1: hidden = silu(x@Wu) * (x@Wg) -----------------------
__global__ __launch_bounds__(256)
void moe_expert_up(const float* __restrict__ x, const float* __restrict__ Wg,
                   const float* __restrict__ Wu, const int* __restrict__ cnt,
                   const int* __restrict__ lists, __bf16* __restrict__ hid) {
  const int e = blockIdx.z;
  const int ce = cnt[e];
  const int my = blockIdx.y;
  if (my * TM >= ce) return;
  const int hbase = blockIdx.x * TN;
  const int tid = threadIdx.x;
  const int lane = tid & 31, wid = tid >> 5;
  const int wm = wid & 3, wn = wid >> 2;

  __shared__ __align__(16) __bf16 sA[2][TM * LDP];
  __shared__ __align__(16) __bf16 sBg[2][TN * LDP];
  __shared__ __align__(16) __bf16 sBu[2][TN * LDP];
  __shared__ int slotArr[TM];

  if (tid < TM) {
    int gm = my * TM + tid;
    slotArr[tid] = (gm < ce) ? lists[e * NT + gm] : -1;
  }
  __syncthreads();

  // per-thread staging coordinates (fixed for the whole K loop)
  const int aM0 = tid >> 3, aQ = tid & 7, aM1 = aM0 + 32;
  const int bKp = tid & 7;                 // k-pair index (rows 2kp, 2kp+1)
  const int bN4 = (tid >> 3) & 31;         // n-group of 4 columns

  const int s0 = slotArr[aM0], s1 = slotArr[aM1];
  const float* xr0 = x + (size_t)((s0 < 0 ? 0 : s0) >> 1) * DIM;
  const float* xr1 = x + (size_t)((s1 < 0 ? 0 : s1) >> 1) * DIM;
  const float m0 = (s0 < 0) ? 0.f : 1.f;
  const float m1 = (s1 < 0) ? 0.f : 1.f;

  const float* wgB = Wg + (size_t)e * DIM * HID + hbase + bN4 * 4;
  const float* wuB = Wu + (size_t)e * DIM * HID + hbase + bN4 * 4;

  float4 ra0, ra1, g0a, g0b, g1a, g1b, u0a, u0b, u1a, u1b;

  auto loadStage = [&](int kt) {   // global -> regs only (no LDS): legal pre-barrier
    const int kb = kt * KS;
    ra0 = LD4(xr0 + kb + aQ * 4);
    ra1 = LD4(xr1 + kb + aQ * 4);
    const float* p0 = wgB + (size_t)(kb + 2 * bKp) * HID;
    g0a = LD4(p0); g0b = LD4(p0 + HID);
    const float* p1 = wgB + (size_t)(kb + 2 * bKp + 16) * HID;
    g1a = LD4(p1); g1b = LD4(p1 + HID);
    const float* q0 = wuB + (size_t)(kb + 2 * bKp) * HID;
    u0a = LD4(q0); u0b = LD4(q0 + HID);
    const float* q1 = wuB + (size_t)(kb + 2 * bKp + 16) * HID;
    u1a = LD4(q1); u1b = LD4(q1 + HID);
  };
  auto storeStage = [&](int b) {
    uint2* pa = (uint2*)&sA[b][0];
    uint2 w0, w1;
    w0.x = pack_bf16(ra0.x * m0, ra0.y * m0);
    w0.y = pack_bf16(ra0.z * m0, ra0.w * m0);
    w1.x = pack_bf16(ra1.x * m1, ra1.y * m1);
    w1.y = pack_bf16(ra1.z * m1, ra1.w * m1);
    pa[aM0 * 10 + aQ] = w0;
    pa[aM1 * 10 + aQ] = w1;
    unsigned* pg = (unsigned*)&sBg[b][0];
    unsigned* pu = (unsigned*)&sBu[b][0];
    const int r0 = bN4 * 4;
    pg[(r0 + 0) * 20 + bKp]     = pack_bf16(g0a.x, g0b.x);
    pg[(r0 + 1) * 20 + bKp]     = pack_bf16(g0a.y, g0b.y);
    pg[(r0 + 2) * 20 + bKp]     = pack_bf16(g0a.z, g0b.z);
    pg[(r0 + 3) * 20 + bKp]     = pack_bf16(g0a.w, g0b.w);
    pg[(r0 + 0) * 20 + bKp + 8] = pack_bf16(g1a.x, g1b.x);
    pg[(r0 + 1) * 20 + bKp + 8] = pack_bf16(g1a.y, g1b.y);
    pg[(r0 + 2) * 20 + bKp + 8] = pack_bf16(g1a.z, g1b.z);
    pg[(r0 + 3) * 20 + bKp + 8] = pack_bf16(g1a.w, g1b.w);
    pu[(r0 + 0) * 20 + bKp]     = pack_bf16(u0a.x, u0b.x);
    pu[(r0 + 1) * 20 + bKp]     = pack_bf16(u0a.y, u0b.y);
    pu[(r0 + 2) * 20 + bKp]     = pack_bf16(u0a.z, u0b.z);
    pu[(r0 + 3) * 20 + bKp]     = pack_bf16(u0a.w, u0b.w);
    pu[(r0 + 0) * 20 + bKp + 8] = pack_bf16(u1a.x, u1b.x);
    pu[(r0 + 1) * 20 + bKp + 8] = pack_bf16(u1a.y, u1b.y);
    pu[(r0 + 2) * 20 + bKp + 8] = pack_bf16(u1a.z, u1b.z);
    pu[(r0 + 3) * 20 + bKp + 8] = pack_bf16(u1a.w, u1b.w);
  };

  v8f accG[4] = {}; v8f accU[4] = {};
  loadStage(0); storeStage(0);
  const int NK = DIM / KS;
  for (int kt = 0; kt < NK; ++kt) {
    const int cur = kt & 1;
    if (kt + 1 < NK) loadStage(kt + 1);   // issue VMEM before the barrier
    __syncthreads();

    const __bf16* ab = &sA[cur][(wm * 16 + (lane & 15)) * LDP + ((lane < 16) ? 0 : 8)];
    v16bf a = cat16(*(const v8bf*)ab, *(const v8bf*)(ab + 16));
#pragma unroll
    for (int nt = 0; nt < 4; ++nt) {
      const int row = (wn * 64 + nt * 16 + (lane & 15)) * LDP + ((lane < 16) ? 0 : 8);
      const __bf16* bg = &sBg[cur][row];
      v16bf b = cat16(*(const v8bf*)bg, *(const v8bf*)(bg + 16));
      accG[nt] = __builtin_amdgcn_wmma_f32_16x16x32_bf16(
          false, a, false, b, (short)0, accG[nt], false, false);
      const __bf16* bu = &sBu[cur][row];
      v16bf c = cat16(*(const v8bf*)bu, *(const v8bf*)(bu + 16));
      accU[nt] = __builtin_amdgcn_wmma_f32_16x16x32_bf16(
          false, a, false, c, (short)0, accU[nt], false, false);
    }
    if (kt + 1 < NK) storeStage(1 - cur);
  }

  const int cRem = ce - my * TM;
#pragma unroll
  for (int nt = 0; nt < 4; ++nt) {
#pragma unroll
    for (int r = 0; r < 8; ++r) {
      int ml = wm * 16 + r + ((lane >> 4) << 3);
      if (ml < cRem) {
        int slot = slotArr[ml];
        float u = accU[nt][r], g = accG[nt][r];
        float hv = g * (u / (1.f + __expf(-u)));   // silu(up) * gate
        int h = hbase + wn * 64 + nt * 16 + (lane & 15);
        hid[(size_t)slot * HID + h] = (__bf16)hv;
      }
    }
  }
}

// ---------------- Phase 2: y[slot] = w[slot] * (hidden @ Wd[e]) ---------------
__global__ __launch_bounds__(256)
void moe_expert_down(const __bf16* __restrict__ hid, const float* __restrict__ Wd,
                     const int* __restrict__ cnt, const int* __restrict__ lists,
                     const float* __restrict__ topW, float* __restrict__ y) {
  const int e = blockIdx.z;
  const int ce = cnt[e];
  const int my = blockIdx.y;
  if (my * TM >= ce) return;
  const int dbase = blockIdx.x * TN;
  const int tid = threadIdx.x;
  const int lane = tid & 31, wid = tid >> 5;
  const int wm = wid & 3, wn = wid >> 2;

  __shared__ __align__(16) __bf16 sA[2][TM * LDP];
  __shared__ __align__(16) __bf16 sB[2][TN * LDP];
  __shared__ int slotArr[TM];
  __shared__ float wArr[TM];

  if (tid < TM) {
    int gm = my * TM + tid;
    int slot = (gm < ce) ? lists[e * NT + gm] : -1;
    slotArr[tid] = slot;
    wArr[tid] = (slot >= 0) ? topW[slot] : 0.f;
  }
  __syncthreads();

  // staging coords
  const int aM = tid >> 2, aQ4 = tid & 3;          // one row per 4 threads, 16B quads
  const int bKp = tid & 7;
  const int bN4 = (tid >> 3) & 31;

  const int sM = slotArr[aM];
  const __bf16* hrow = hid + (size_t)(sM < 0 ? 0 : sM) * HID;
  const bool av = (sM >= 0);

  const float* wdB = Wd + (size_t)e * HID * DIM + dbase + bN4 * 4;

  uint4 rA;
  float4 b0a, b0b, b1a, b1b;

  auto loadStage = [&](int kt) {   // global -> regs only (no LDS): legal pre-barrier
    const int kb = kt * KS;
    rA = *(const uint4*)(hrow + kb + aQ4 * 8);
    const float* p0 = wdB + (size_t)(kb + 2 * bKp) * DIM;
    b0a = LD4(p0); b0b = LD4(p0 + DIM);
    const float* p1 = wdB + (size_t)(kb + 2 * bKp + 16) * DIM;
    b1a = LD4(p1); b1b = LD4(p1 + DIM);
  };
  auto storeStage = [&](int b) {
    uint4 w = rA;
    w.x = av ? w.x : 0u; w.y = av ? w.y : 0u;
    w.z = av ? w.z : 0u; w.w = av ? w.w : 0u;
    ((uint4*)&sA[b][0])[aM * 5 + aQ4] = w;
    unsigned* pb = (unsigned*)&sB[b][0];
    const int r0 = bN4 * 4;
    pb[(r0 + 0) * 20 + bKp]     = pack_bf16(b0a.x, b0b.x);
    pb[(r0 + 1) * 20 + bKp]     = pack_bf16(b0a.y, b0b.y);
    pb[(r0 + 2) * 20 + bKp]     = pack_bf16(b0a.z, b0b.z);
    pb[(r0 + 3) * 20 + bKp]     = pack_bf16(b0a.w, b0b.w);
    pb[(r0 + 0) * 20 + bKp + 8] = pack_bf16(b1a.x, b1b.x);
    pb[(r0 + 1) * 20 + bKp + 8] = pack_bf16(b1a.y, b1b.y);
    pb[(r0 + 2) * 20 + bKp + 8] = pack_bf16(b1a.z, b1b.z);
    pb[(r0 + 3) * 20 + bKp + 8] = pack_bf16(b1a.w, b1b.w);
  };

  v8f acc[4] = {};
  loadStage(0); storeStage(0);
  const int NK = HID / KS;
  for (int kt = 0; kt < NK; ++kt) {
    const int cur = kt & 1;
    if (kt + 1 < NK) loadStage(kt + 1);   // issue VMEM before the barrier
    __syncthreads();

    const __bf16* ab = &sA[cur][(wm * 16 + (lane & 15)) * LDP + ((lane < 16) ? 0 : 8)];
    v16bf a = cat16(*(const v8bf*)ab, *(const v8bf*)(ab + 16));
#pragma unroll
    for (int nt = 0; nt < 4; ++nt) {
      const __bf16* bb = &sB[cur][(wn * 64 + nt * 16 + (lane & 15)) * LDP + ((lane < 16) ? 0 : 8)];
      v16bf b = cat16(*(const v8bf*)bb, *(const v8bf*)(bb + 16));
      acc[nt] = __builtin_amdgcn_wmma_f32_16x16x32_bf16(
          false, a, false, b, (short)0, acc[nt], false, false);
    }
    if (kt + 1 < NK) storeStage(1 - cur);
  }

  const int cRem = ce - my * TM;
#pragma unroll
  for (int nt = 0; nt < 4; ++nt) {
#pragma unroll
    for (int r = 0; r < 8; ++r) {
      int ml = wm * 16 + r + ((lane >> 4) << 3);
      if (ml < cRem) {
        int slot = slotArr[ml];
        int d = dbase + wn * 64 + nt * 16 + (lane & 15);
        y[(size_t)slot * DIM + d] = wArr[ml] * acc[nt][r];
      }
    }
  }
}

// ---------------- Combine + finalize ------------------------------------------
__global__ __launch_bounds__(256)
void moe_combine(const float* __restrict__ y, float* __restrict__ out) {
  size_t q = (size_t)blockIdx.x * 256 + threadIdx.x;     // one float4 per thread
  size_t t = q >> 8; int d4 = (int)(q & 255);            // 256 quads per token row
  const float4 a = *(const float4*)(y + (t * 2) * DIM + d4 * 4);
  const float4 b = *(const float4*)(y + (t * 2 + 1) * DIM + d4 * 4);
  float4 r; r.x = a.x + b.x; r.y = a.y + b.y; r.z = a.z + b.z; r.w = a.w + b.w;
  *(float4*)(out + q * 4) = r;
}

__global__ void moe_finalize(const float* __restrict__ blockImp,
                             const int* __restrict__ blockCnt,
                             const float* __restrict__ blockEnt,
                             float* __restrict__ outTail) {
  if (threadIdx.x != 0 || blockIdx.x != 0) return;
  float imp[NE], cf[NE], ent = 0.f;
  for (int e = 0; e < NE; ++e) { imp[e] = 0.f; cf[e] = 0.f; }
  for (int b = 0; b < NT / 256; ++b) {
    for (int e = 0; e < NE; ++e) {
      imp[e] += blockImp[b * NE + e];
      cf[e] += (float)blockCnt[b * NE + e];
    }
    ent += blockEnt[b];
  }
  float aux = 0.f;
  for (int e = 0; e < NE; ++e)
    aux += (imp[e] / (float)NT) * (cf[e] / ((float)NT + 1e-6f));
  outTail[0] = aux * (float)NE * 0.01f;
  outTail[1] = (ent / (float)NT) * 0.01f;
  outTail[2] = 0.f;
}

extern "C" void kernel_launch(void* const* d_in, const int* in_sizes, int n_in,
                              void* d_out, int out_size, void* d_ws, size_t ws_size,
                              hipStream_t stream) {
  const float* x    = (const float*)d_in[0];
  const float* rw   = (const float*)d_in[1];
  const float* temp = (const float*)d_in[2];
  const float* Wg   = (const float*)d_in[3];
  const float* Wu   = (const float*)d_in[4];
  const float* Wd   = (const float*)d_in[5];
  float* out = (float*)d_out;

  char* ws = (char*)d_ws;
  size_t off = 0;
  auto alloc = [&](size_t bytes) {
    size_t o = off;
    off = (off + bytes + 255) & ~(size_t)255;
    return o;
  };
  int*    cnt      = (int*)   (ws + alloc(NE * sizeof(int)));
  float*  blockImp = (float*) (ws + alloc((NT / 256) * NE * sizeof(float)));
  int*    blockCnt = (int*)   (ws + alloc((NT / 256) * NE * sizeof(int)));
  float*  blockEnt = (float*) (ws + alloc((NT / 256) * sizeof(float)));
  int*    topI     = (int*)   (ws + alloc(2 * NT * sizeof(int)));
  float*  topW     = (float*) (ws + alloc(2 * NT * sizeof(float)));
  int*    lists    = (int*)   (ws + alloc((size_t)NE * NT * sizeof(int)));
  __bf16* hid      = (__bf16*)(ws + alloc((size_t)2 * NT * HID * sizeof(__bf16)));
  float*  y        = (float*) (ws + alloc((size_t)2 * NT * DIM * sizeof(float)));
  (void)ws_size; (void)in_sizes; (void)n_in; (void)out_size;

  hipMemsetAsync(cnt, 0, NE * sizeof(int), stream);

  moe_router<<<NT / 256, 256, 0, stream>>>(x, rw, temp, topI, topW,
                                           blockImp, blockCnt, blockEnt);
  moe_dispatch<<<NT / 256, 256, 0, stream>>>(topI, cnt, lists);

  dim3 g1(HID / TN, NT / TM, NE);   // 32 x 128 x 8 (blocks early-exit past count)
  moe_expert_up<<<g1, 256, 0, stream>>>(x, Wg, Wu, cnt, lists, hid);

  dim3 g2(DIM / TN, NT / TM, NE);   // 8 x 128 x 8
  moe_expert_down<<<g2, 256, 0, stream>>>(hid, Wd, cnt, lists, topW, y);

  moe_combine<<<(NT * DIM / 4) / 256, 256, 0, stream>>>(y, out);
  moe_finalize<<<1, 32, 0, stream>>>(blockImp, blockCnt, blockEnt,
                                     out + (size_t)NT * DIM);
}
